// BasicBlock_37228776522110
// MI455X (gfx1250) — 8-bit matrix operand order repaired and checked by reading, compiled, not run on hardware
//
#include <hip/hip_runtime.h>
#include <hip/hip_bf16.h>

typedef int v8i __attribute__((ext_vector_type(8)));
typedef int v4i __attribute__((ext_vector_type(4)));

#define BB   64
#define CC   64
#define PP   64
#define HH   56
#define WW   56
#define PH   58
#define PW   66
#define KTAP 9

#define A_BYTES   ((size_t)BB * PH * PW * CC)
#define B_BYTES   ((size_t)4 * KTAP * 32 * 32)
#define C_OFF     (A_BYTES + B_BYTES)

__global__ __launch_bounds__(256) void zero_ws_kernel(int4* p, int n16) {
    int i = blockIdx.x * 256 + threadIdx.x;
    if (i < n16) p[i] = make_int4(0, 0, 0, 0);
}

__global__ __launch_bounds__(256) void pack_act_kernel(const float* __restrict__ x,
                                                       const float* __restrict__ bias0,
                                                       signed char* __restrict__ A) {
    int pix = blockIdx.x * 256 + threadIdx.x;
    if (pix >= BB * HH * WW) return;
    int w = pix % WW;
    int h = (pix / WW) % HH;
    int b = pix / (WW * HH);
    size_t base = ((size_t)(b * PH + (h + 1)) * PW + (w + 1)) * CC;
    #pragma unroll
    for (int i = 0; i < 64; ++i) {
        int hf = i >> 5, r = i & 31, v = r >> 2, j = r & 3;
        int c = hf * 16 + (v & 3) * 4 + (v >> 2) * 32 + j;
        float val = x[((size_t)(b * CC + c) * HH + h) * WW + w] + bias0[c];
        A[base + i] = (signed char)((val > 0.0f) - (val < 0.0f));
    }
}

__global__ void pack_w_kernel(const float* __restrict__ w, const float* __restrict__ gamma,
                              const float* __restrict__ beta, const float* __restrict__ rmean,
                              const float* __restrict__ rvar, const float* __restrict__ bias1,
                              const float* __restrict__ alpha, const float* __restrict__ bias2,
                              signed char* __restrict__ Bw, float* __restrict__ consts) {
    int p = threadIdx.x;
    if (p >= PP) return;
    float s = 0.0f;
    for (int i = 0; i < CC * KTAP; ++i) s += fabsf(w[p * CC * KTAP + i]);
    s *= (1.0f / (CC * KTAP));
    float inv = gamma[p] * rsqrtf(rvar[p] + 1e-5f);
    consts[p]       = s * inv;
    consts[64 + p]  = beta[p] - rmean[p] * inv + bias1[p];
    consts[128 + p] = alpha[p];
    consts[192 + p] = bias2[p];

    int nt = p >> 4, lm = p & 15;
    for (int t = 0; t < KTAP; ++t)
        for (int hf = 0; hf < 2; ++hf)
            for (int v = 0; v < 8; ++v)
                for (int j = 0; j < 4; ++j) {
                    int K = (v < 4) ? (hf * 16 + v * 4 + j)
                                    : (32 + hf * 16 + (v - 4) * 4 + j);
                    float wv = w[(size_t)(p * CC + K) * KTAP + t];
                    size_t off = ((size_t)((nt * KTAP + t) * 32) + (hf * 16 + lm)) * 32 + v * 4 + j;
                    Bw[off] = (signed char)((wv > 0.0f) - (wv < 0.0f));
                }
}

__global__ __launch_bounds__(256) void conv_wmma_kernel(const signed char* __restrict__ A,
                                                        const signed char* __restrict__ Bw,
                                                        const float* __restrict__ consts,
                                                        const float* __restrict__ xres,
                                                        float* __restrict__ out) {
    int wave = threadIdx.x >> 5;
    int lane = threadIdx.x & 31;
    int tile = blockIdx.x * 8 + wave;
    int wt = tile & 3;
    int h  = (tile >> 2) % HH;
    int b  = (tile >> 2) / HH;
    int w0 = wt * 16;
    int r    = lane & 15;
    int half = lane >> 4;

    v8i acc[4] = {};
    #pragma unroll
    for (int t = 0; t < KTAP; ++t) {
        int dh = t / 3, dw = t % 3;
        size_t apix = ((size_t)(b * PH + (h + dh)) * PW + (w0 + r + dw)) * CC + (size_t)half * 32;
        v4i a0 = *(const v4i*)(A + apix);
        v4i a1 = *(const v4i*)(A + apix + 16);
        v8i av = { a0[0], a0[1], a0[2], a0[3], a1[0], a1[1], a1[2], a1[3] };
        #pragma unroll
        for (int nt = 0; nt < 4; ++nt) {
            size_t boff = ((size_t)((nt * KTAP + t) * 32) + lane) * 32;
            v4i b0 = *(const v4i*)(Bw + boff);
            v4i b1 = *(const v4i*)(Bw + boff + 16);
            v8i bv = { b0[0], b0[1], b0[2], b0[3], b1[0], b1[1], b1[2], b1[3] };
            acc[nt] = __builtin_amdgcn_wmma_i32_16x16x64_iu8(
                true, av, true, bv, acc[nt], false, false);
        }
    }

    const float* m  = consts;
    const float* c0 = consts + 64;
    const float* al = consts + 128;
    const float* b2 = consts + 192;
    #pragma unroll
    for (int nt = 0; nt < 4; ++nt) {
        int p = nt * 16 + (lane & 15);
        float mm = m[p], cc = c0[p], aa = al[p], bb = b2[p];
        #pragma unroll
        for (int k = 0; k < 8; ++k) {
            int M = half * 8 + k;
            int w = w0 + M;
            if (w < WW) {
                size_t oidx = ((size_t)(b * PP + p) * HH + h) * WW + w;
                float v = (float)acc[nt][k] * mm + xres[oidx] + cc;
                v = (v >= 0.0f) ? v : aa * v;
                out[oidx] = v + bb;
            }
        }
    }
}

extern "C" void kernel_launch(void* const* d_in, const int* in_sizes, int n_in,
                              void* d_out, int out_size, void* d_ws, size_t ws_size,
                              hipStream_t stream) {
    const float* x     = (const float*)d_in[0];
    const float* bias0 = (const float*)d_in[1];
    const float* w     = (const float*)d_in[2];
    const float* gamma = (const float*)d_in[3];
    const float* beta  = (const float*)d_in[4];
    const float* rmean = (const float*)d_in[5];
    const float* rvar  = (const float*)d_in[6];
    const float* bias1 = (const float*)d_in[7];
    const float* alpha = (const float*)d_in[8];
    const float* bias2 = (const float*)d_in[9];

    signed char* Apk   = (signed char*)d_ws;
    signed char* Bpk   = Apk + A_BYTES;
    float*       cns   = (float*)((char*)d_ws + C_OFF);
    float*       out   = (float*)d_out;

    int n16 = (int)(A_BYTES / 16);
    zero_ws_kernel<<<(n16 + 255) / 256, 256, 0, stream>>>((int4*)Apk, n16);

    int npix = BB * HH * WW;
    pack_act_kernel<<<(npix + 255) / 256, 256, 0, stream>>>(x, bias0, Apk);

    pack_w_kernel<<<1, 64, 0, stream>>>(w, gamma, beta, rmean, rvar, bias1, alpha, bias2,
                                        Bpk, cns);

    conv_wmma_kernel<<<14336 / 8, 256, 0, stream>>>(Apk, Bpk, cns, x, out);
}
